// VQVAE_8899172237626
// MI455X (gfx1250) — compile-verified, matrix operand order checked by reading, not run on hardware
//
#include <hip/hip_runtime.h>
#include <hip/hip_bf16.h>

typedef __attribute__((ext_vector_type(16))) __bf16 v16bf;
typedef __attribute__((ext_vector_type(8)))  __bf16 v8bf;
typedef __attribute__((ext_vector_type(8)))  float  v8f;
typedef unsigned int v4u __attribute__((ext_vector_type(4)));
typedef int v8i __attribute__((ext_vector_type(8)));
typedef int v4i __attribute__((ext_vector_type(4)));

union Frag { v16bf v; struct { v8bf lo, hi; } p; };

__device__ __forceinline__ float bf2f(__bf16 v) {
    unsigned short s = __builtin_bit_cast(unsigned short, v);
    unsigned int u = ((unsigned int)s) << 16;
    return __builtin_bit_cast(float, u);
}
__device__ __forceinline__ __bf16 f2bf(float f) {
    unsigned int u = __builtin_bit_cast(unsigned int, f);
    u += 0x7FFFu + ((u >> 16) & 1u);           // round-to-nearest-even
    unsigned short s = (unsigned short)(u >> 16);
    return __builtin_bit_cast(__bf16, s);
}
constexpr int ilog2c(int v) { return v <= 1 ? 0 : 1 + ilog2c(v >> 1); }

// ---------------------------------------------------------------------------
// TDM: DMA a [rows x widthElems] bf16 panel (row stride strideElems) into LDS.
// D# per CDNA5 ISA ch.8 (6-arg builtin form on this toolchain).
// ---------------------------------------------------------------------------
__device__ __forceinline__ void tdm_load_panel(const __bf16* gsrc, unsigned ldsOff,
                                               int width, int rows, int strideElems)
{
    unsigned long long ga = (unsigned long long)gsrc;
    v4u g0 = {0u, 0u, 0u, 0u};
    g0[0] = 1u;                                              // count=1 (valid D#)
    g0[1] = ldsOff;                                          // lds_addr (bytes)
    g0[2] = (unsigned)ga;                                    // global_addr[31:0]
    g0[3] = (unsigned)((ga >> 32) & 0x1FFFFFFu) | (2u << 30);// [56:32] | type=2
    v8i g1 = {0, 0, 0, 0, 0, 0, 0, 0};
    g1[0] = 1 << 16;                                         // data_size=1 (2 bytes)
    g1[1] = (width & 0xFFFF) << 16;                          // tensor_dim0[15:0]
    g1[2] = ((rows & 0xFFFF) << 16);                         // tdim0 hi | tensor_dim1 lo
    g1[3] = (width << 16);                                   // tdim1 hi | tile_dim0
    g1[4] = rows & 0xFFFF;                                   // tile_dim1 (tile_dim2=0)
    g1[5] = strideElems;                                     // tensor_dim0_stride[31:0]
    v4i gz4 = {0, 0, 0, 0};
    v8i gz8 = {0, 0, 0, 0, 0, 0, 0, 0};
    __builtin_amdgcn_tensor_load_to_lds(g0, g1, gz4, gz4, gz8, 0);
}

// ---------------------------------------------------------------------------
// Implicit-GEMM conv, NHWC bf16 activations, K-order (kh,kw,ci), weights in
// fragment-ordered layout [(ktile,hi)][col][j] so each lane's B fragment is a
// contiguous 32B LDS chunk (2x ds_load_b128) and, when CIN%32==0, each lane's
// A fragment is two contiguous b128 global loads with one bounds check per
// K-tile. One wave = 64(pixels) x 16(channels), 4 WMMAs per K-step.
// ACT: 0=none 1=leaky 2=relu 3=sigmoid+nan_to_num (TOF32 writes NCHW fp32)
// ---------------------------------------------------------------------------
template<int CIN,int HIN,int WIN,int COUT,int COP,int HOUT,int WOUT,
         int KH,int KW,int STR,int PAD,int LDIL,int ACT,bool HASB,bool HASR,bool TOF32>
__global__ __launch_bounds__(128)
void conv_tmpl(const __bf16* __restrict__ X, const __bf16* __restrict__ Wt,
               const float* __restrict__ bias, const __bf16* __restrict__ Rsd,
               __bf16* __restrict__ Y, float* __restrict__ Yf)
{
    constexpr int  KD    = CIN * KH * KW;
    constexpr int  KP    = (KD + 31) & ~31;
    constexpr int  HW    = HOUT * WOUT;
    constexpr int  LW    = ilog2c(WOUT);
    constexpr int  LHW   = ilog2c(HW);
    constexpr int  HD    = LDIL * (HIN - 1) + 1;
    constexpr int  WD    = LDIL * (WIN - 1) + 1;
    constexpr bool FASTA = (CIN % 32) == 0;

    extern __shared__ __bf16 sB[];                 // [KP/16 rows][16 cols][16 j]

    const int lane = threadIdx.x;
    const int wv   = threadIdx.y;
    const int l15  = lane & 15;
    const int hi   = lane >> 4;

    if (wv == 0) {                                 // one TDM per workgroup
        tdm_load_panel(Wt + (size_t)blockIdx.y * 256,
                       (unsigned)(unsigned long long)(void*)&sB[0],
                       256, KP / 16, COP * 16);
        __builtin_amdgcn_s_wait_tensorcnt(0);
    }
    __syncthreads();

    const int mBase = (blockIdx.x * 4 + wv) * 64;
    const int coCol = blockIdx.y * 16 + l15;

    int an[4], aoh[4], aow[4];
#pragma unroll
    for (int r = 0; r < 4; r++) {
        int m = mBase + r * 16 + l15;
        an[r] = m >> LHW; aoh[r] = (m >> LW) & (HOUT - 1); aow[r] = m & (WOUT - 1);
    }

    const v8f vzero = {0.f,0.f,0.f,0.f,0.f,0.f,0.f,0.f};
    v8f acc[4];
#pragma unroll
    for (int r = 0; r < 4; r++) acc[r] = vzero;
    const __bf16 bz = __builtin_bit_cast(__bf16, (unsigned short)0);
    const v8bf z8 = {bz, bz, bz, bz, bz, bz, bz, bz};

    for (int k0 = 0; k0 < KP; k0 += 32) {
        const int ktile = k0 >> 5;
        const v16bf bfrag = *(const v16bf*)&sB[((ktile * 2 + hi) * 16 + l15) * 16];

        v16bf afrag[4];
        if constexpr (FASTA) {
            const int win   = k0 >> ilog2c(CIN);
            const int cbase = k0 & (CIN - 1);
            const int kh = win / KW, kw = win - (win / KW) * KW;
#pragma unroll
            for (int r = 0; r < 4; r++) {
                int ihd = aoh[r] * STR - PAD + kh;
                int iwd = aow[r] * STR - PAD + kw;
                bool ok = true;
                if constexpr (PAD > 0 || LDIL > 1) {
                    ok = (ihd >= 0) && (ihd < HD) && (iwd >= 0) && (iwd < WD);
                    if constexpr (LDIL > 1)
                        ok = ok && ((ihd & (LDIL - 1)) == 0) && ((iwd & (LDIL - 1)) == 0);
                }
                Frag u;
                if (ok) {
                    const int ih = (LDIL > 1) ? (ihd >> ilog2c(LDIL)) : ihd;
                    const int iw = (LDIL > 1) ? (iwd >> ilog2c(LDIL)) : iwd;
                    const __bf16* src =
                        &X[((size_t)(an[r] * HIN + ih) * WIN + iw) * CIN + cbase + hi * 8];
                    u.p.lo = *(const v8bf*)(src);          // j 0..7  -> K +hi*8
                    u.p.hi = *(const v8bf*)(src + 16);     // j 8..15 -> K +16+hi*8
                } else { u.p.lo = z8; u.p.hi = z8; }
                afrag[r] = u.v;
            }
        } else {                                   // scalar gather (stem, CIN=3)
#pragma unroll
            for (int j = 0; j < 16; j++) {
                int kk = k0 + ((j < 8) ? j : j + 8) + hi * 8;
                bool kv = (kk < KD);
                int win = kk / CIN, ci = kk - (kk / CIN) * CIN;
                int kh = win / KW, kw = win - (win / KW) * KW;
#pragma unroll
                for (int r = 0; r < 4; r++) {
                    __bf16 v = bz;
                    int ihd = aoh[r] * STR - PAD + kh;
                    int iwd = aow[r] * STR - PAD + kw;
                    bool ok = kv && (ihd >= 0) && (ihd < HD) && (iwd >= 0) && (iwd < WD);
                    if (ok)
                        v = X[((size_t)(an[r] * HIN + ihd) * WIN + iwd) * CIN + ci];
                    afrag[r][j] = v;
                }
            }
        }
#pragma unroll
        for (int r = 0; r < 4; r++)
            acc[r] = __builtin_amdgcn_wmma_f32_16x16x32_bf16(
                         false, afrag[r], false, bfrag, (short)0, acc[r], false, false);
    }

    if constexpr (COUT < COP) { if (coCol >= COUT) return; }
    float bv = 0.f;
    if constexpr (HASB) bv = bias[coCol];
#pragma unroll
    for (int r = 0; r < 4; r++) {
#pragma unroll
        for (int v = 0; v < 8; v++) {
            int m = mBase + r * 16 + v + hi * 8;           // C row = v + 8*hi
            float val = acc[r][v] + bv;
            if constexpr (HASR) val += bf2f(Rsd[(size_t)m * COUT + coCol]);
            if constexpr (ACT == 1) val = val > 0.f ? val : 0.01f * val;
            else if constexpr (ACT == 2) val = val > 0.f ? val : 0.f;
            else if constexpr (ACT == 3) { val = 1.f / (1.f + __expf(-val)); if (val != val) val = 0.f; }
            if constexpr (TOF32) {                         // NCHW fp32 (final output)
                int n = m >> LHW; int oh = (m >> LW) & (HOUT - 1); int ow = m & (WOUT - 1);
                Yf[(((size_t)n * COUT + coCol) * HOUT + oh) * WOUT + ow] = val;
            } else {                                       // NHWC bf16
                Y[(size_t)m * COUT + coCol] = f2bf(val);
            }
        }
    }
}

// ---------------------------------------------------------------------------
// VQ argmin: dist = |e|^2 - 2*dot via WMMA; E is NHWC (= the flat matrix),
// codebook packed fragment-ordered [(t,hi)][code][j]. shfl_xor row reduction.
// ---------------------------------------------------------------------------
__global__ __launch_bounds__(128)
void vq_argmin_kernel(const __bf16* __restrict__ E, const __bf16* __restrict__ Epk,
                      const float* __restrict__ enorm, int* __restrict__ idxOut)
{
    const int lane = threadIdx.x, wv = threadIdx.y;
    const int tile = blockIdx.x * 4 + wv;
    const int l15 = lane & 15, hi = lane >> 4;
    const int rowA = tile * 16 + l15;

    Frag u0, u1;
    const __bf16* src = &E[(size_t)rowA * 64 + hi * 8];
    u0.p.lo = *(const v8bf*)(src);
    u0.p.hi = *(const v8bf*)(src + 16);
    u1.p.lo = *(const v8bf*)(src + 32);
    u1.p.hi = *(const v8bf*)(src + 48);
    v16bf af[2] = {u0.v, u1.v};

    float best[8]; int bidx[8];
#pragma unroll
    for (int v = 0; v < 8; v++) { best[v] = 3.4e38f; bidx[v] = 0; }

    const v8f vzero = {0.f,0.f,0.f,0.f,0.f,0.f,0.f,0.f};
    for (int ct = 0; ct < 32; ct++) {
        const int col = ct * 16 + l15;
        v8f acc = vzero;
#pragma unroll
        for (int t = 0; t < 2; t++) {
            const v16bf bf = *(const v16bf*)&Epk[(((size_t)(t * 2 + hi) * 512) + col) * 16];
            acc = __builtin_amdgcn_wmma_f32_16x16x32_bf16(
                      false, af[t], false, bf, (short)0, acc, false, false);
        }
        float en = enorm[col];
#pragma unroll
        for (int v = 0; v < 8; v++) {
            float d = en - 2.f * acc[v];
            if (d < best[v] || (d == best[v] && col < bidx[v])) { best[v] = d; bidx[v] = col; }
        }
    }
#pragma unroll
    for (int off = 1; off < 16; off <<= 1) {
#pragma unroll
        for (int v = 0; v < 8; v++) {
            float od = __shfl_xor(best[v], off, 32);
            int   oi = __shfl_xor(bidx[v], off, 32);
            if (od < best[v] || (od == best[v] && oi < bidx[v])) { best[v] = od; bidx[v] = oi; }
        }
    }
    if (l15 == 0) {
#pragma unroll
        for (int v = 0; v < 8; v++)
            idxOut[tile * 16 + v + hi * 8] = bidx[v];
    }
}

// quant = emb[idx] (straight-through forward value); accumulate (1+beta)*MSE.
__global__ void vq_gather_kernel(const int* __restrict__ idx, const float* __restrict__ emb,
                                 const __bf16* __restrict__ E, __bf16* __restrict__ Q,
                                 float* __restrict__ accum)
{
    __shared__ float red[256];
    int p = blockIdx.x * 256 + threadIdx.x;
    float s = 0.f;
    if (p < 32768) {
        int id = idx[p];
        for (int c = 0; c < 64; c++) {
            float e = emb[id * 64 + c];
            size_t o = (size_t)p * 64 + c;                 // NHWC
            float d = e - bf2f(E[o]);
            Q[o] = f2bf(e);
            s += d * d;
        }
    }
    red[threadIdx.x] = s;
    __syncthreads();
    for (int st = 128; st > 0; st >>= 1) {
        if (threadIdx.x < st) red[threadIdx.x] += red[threadIdx.x + st];
        __syncthreads();
    }
    if (threadIdx.x == 0) atomicAdd(accum, red[0]);
}

__global__ void vq_loss_kernel(const float* __restrict__ acc, float* __restrict__ out)
{
    if (threadIdx.x == 0) out[0] = 1.25f * acc[0] / 2097152.0f;   // (1+beta)*mean
}

// fp32 OIHW -> bf16 fragment-ordered [(ktile,hi)][co][j], K-order (kh,kw,ci).
__global__ void pack_w_kernel(const float* __restrict__ W, __bf16* __restrict__ out,
                              int Co, int Ci, int KH, int KW, int Kdim, int Kpad,
                              int CoP, int lcop)
{
    int i = blockIdx.x * 256 + threadIdx.x;
    if (i >= Kpad * CoP) return;
    int j = i & 15;
    int t = i >> 4;
    int co = t & (CoP - 1);
    int row = t >> lcop;
    int hi = row & 1, ktile = row >> 1;
    int kk = ktile * 32 + ((j < 8) ? j : j + 8) + hi * 8;
    float v = 0.f;
    if (kk < Kdim && co < Co) {
        int win = kk / Ci; int ci = kk - win * Ci;
        int kh = win / KW; int kw = win - kh * KW;
        v = W[((size_t)(co * Ci + ci) * KH + kh) * KW + kw];
    }
    out[i] = f2bf(v);
}

// emb [512][64] fp32 -> bf16 fragment-ordered [(t,hi)][code][j]
__global__ void pack_emb_kernel(const float* __restrict__ emb, __bf16* __restrict__ out)
{
    int i = blockIdx.x * 256 + threadIdx.x;
    if (i >= 32768) return;
    int j = i & 15;
    int t2 = i >> 4;
    int n = t2 & 511;
    int row = t2 >> 9;
    int hi = row & 1, kt = row >> 1;
    int c = kt * 32 + ((j < 8) ? j : j + 8) + hi * 8;
    out[i] = f2bf(emb[n * 64 + c]);
}

__global__ void enorm_kernel(const float* __restrict__ emb, float* __restrict__ en)
{
    int n = blockIdx.x * 256 + threadIdx.x;
    if (n < 512) { float s = 0.f; for (int k = 0; k < 64; k++) { float e = emb[n * 64 + k]; s += e * e; } en[n] = s; }
}

// x: NCHW fp32 (32,3,128,128) -> NHWC bf16
__global__ void cvt_x_kernel(const float* __restrict__ in, __bf16* __restrict__ out)
{
    int i = blockIdx.x * 256 + threadIdx.x;
    if (i >= 1572864) return;
    int c = i % 3; int p = i / 3;
    int n = p >> 14; int hw = p & 16383;
    out[i] = f2bf(in[((size_t)(n * 3 + c) << 14) + hw]);
}

__global__ void leaky_kernel(__bf16* __restrict__ a, int n)
{
    int i = blockIdx.x * 256 + threadIdx.x;
    if (i < n) { float v = bf2f(a[i]); v = v > 0.f ? v : 0.01f * v; a[i] = f2bf(v); }
}

// ---------------------------------------------------------------------------
extern "C" void kernel_launch(void* const* d_in, const int* in_sizes, int n_in,
                              void* d_out, int out_size, void* d_ws, size_t ws_size,
                              hipStream_t stream)
{
    (void)in_sizes; (void)n_in; (void)out_size; (void)ws_size;
    char* ws = (char*)d_ws;
    size_t off = 0;
    auto balloc = [&](size_t bytes) -> size_t { size_t o = off; off = (off + bytes + 255) & ~(size_t)255; return o; };
    auto bfp = [&](size_t o) -> __bf16* { return (__bf16*)(ws + o); };

    size_t oX   = balloc((size_t)1572864 * 2);   // x NHWC bf16
    size_t oBig = balloc((size_t)16777216 * 2);  // 32x64x64x128 NHWC
    size_t oB1  = balloc((size_t)8388608 * 2);   // 32x32x32x256 NHWC ping
    size_t oB2  = balloc((size_t)8388608 * 2);
    size_t oB3  = balloc((size_t)8388608 * 2);
    size_t oE   = balloc((size_t)2097152 * 2);   // enc out NHWC = flat [32768][64]
    size_t oQ   = balloc((size_t)2097152 * 2);
    size_t oIdx = balloc((size_t)32768 * 4);
    size_t oAcc = balloc(256);
    size_t oEn  = balloc(512 * 4);
    size_t wEnc0 = balloc((size_t)64 * 128 * 2);
    size_t wEnc1 = balloc((size_t)2048 * 256 * 2);
    size_t wEr1[6], wEr2[6], wDr1[6], wDr2[6];
    for (int i = 0; i < 6; i++) { wEr1[i] = balloc((size_t)2304 * 256 * 2); wEr2[i] = balloc((size_t)256 * 256 * 2); }
    size_t wEnc2 = balloc((size_t)256 * 64 * 2);
    size_t wEmbP = balloc((size_t)64 * 512 * 2);
    size_t wDec0 = balloc((size_t)576 * 256 * 2);
    for (int i = 0; i < 6; i++) { wDr1[i] = balloc((size_t)2304 * 256 * 2); wDr2[i] = balloc((size_t)256 * 256 * 2); }
    size_t wT0 = balloc((size_t)4096 * 128 * 2);
    size_t wT1 = balloc((size_t)2048 * 16 * 2);

    auto pack = [&](int inIdx, size_t o, int Co, int Ci, int KH, int KW, int CoP, int lcop) {
        int Kd = Ci * KH * KW, Kp = (Kd + 31) & ~31; int tot = Kp * CoP;
        pack_w_kernel<<<(tot + 255) / 256, 256, 0, stream>>>((const float*)d_in[inIdx], bfp(o),
                                                             Co, Ci, KH, KW, Kd, Kp, CoP, lcop);
    };
    pack(1, wEnc0, 128, 3, 4, 4, 128, 7);
    pack(3, wEnc1, 256, 128, 4, 4, 256, 8);
    for (int i = 0; i < 6; i++) { pack(5 + 2 * i, wEr1[i], 256, 256, 3, 3, 256, 8); pack(6 + 2 * i, wEr2[i], 256, 256, 1, 1, 256, 8); }
    pack(17, wEnc2, 64, 256, 1, 1, 64, 6);
    pack_emb_kernel<<<128, 256, 0, stream>>>((const float*)d_in[19], bfp(wEmbP));
    enorm_kernel<<<2, 256, 0, stream>>>((const float*)d_in[19], (float*)(ws + oEn));
    pack(20, wDec0, 256, 64, 3, 3, 256, 8);
    for (int i = 0; i < 6; i++) { pack(22 + 2 * i, wDr1[i], 256, 256, 3, 3, 256, 8); pack(23 + 2 * i, wDr2[i], 256, 256, 1, 1, 256, 8); }
    pack(34, wT0, 128, 256, 4, 4, 128, 7);
    pack(36, wT1, 3, 128, 4, 4, 16, 4);

    cvt_x_kernel<<<(1572864 + 255) / 256, 256, 0, stream>>>((const float*)d_in[0], bfp(oX));
    (void)hipMemsetAsync(ws + oAcc, 0, sizeof(float), stream);

    const dim3 blk(32, 4);

    // encoder
    conv_tmpl<3,128,128,128,128,64,64,4,4,2,1,1,1,true,false,false>
        <<<dim3(512, 8), blk, 64 * 32, stream>>>(bfp(oX), bfp(wEnc0), (const float*)d_in[2], nullptr, bfp(oBig), nullptr);
    conv_tmpl<128,64,64,256,256,32,32,4,4,2,1,1,1,true,false,false>
        <<<dim3(128, 16), blk, 2048 * 32, stream>>>(bfp(oBig), bfp(wEnc1), (const float*)d_in[4], nullptr, bfp(oB1), nullptr);

    size_t P = oB1, T = oB2, O = oB3;
    for (int i = 0; i < 6; i++) {
        conv_tmpl<256,32,32,256,256,32,32,3,3,1,1,1,2,false,false,false>
            <<<dim3(128, 16), blk, 2304 * 32, stream>>>(bfp(P), bfp(wEr1[i]), nullptr, nullptr, bfp(T), nullptr);
        conv_tmpl<256,32,32,256,256,32,32,1,1,1,0,1,0,false,true,false>
            <<<dim3(128, 16), blk, 256 * 32, stream>>>(bfp(T), bfp(wEr2[i]), nullptr, bfp(P), bfp(O), nullptr);
        size_t t2 = P; P = O; O = t2;
    }
    leaky_kernel<<<(8388608 + 255) / 256, 256, 0, stream>>>(bfp(P), 8388608);
    conv_tmpl<256,32,32,64,64,32,32,1,1,1,0,1,1,true,false,false>
        <<<dim3(128, 4), blk, 256 * 32, stream>>>(bfp(P), bfp(wEnc2), (const float*)d_in[18], nullptr, bfp(oE), nullptr);

    // vector quantization
    vq_argmin_kernel<<<dim3(512), blk, 0, stream>>>(bfp(oE), bfp(wEmbP), (float*)(ws + oEn), (int*)(ws + oIdx));
    vq_gather_kernel<<<128, 256, 0, stream>>>((int*)(ws + oIdx), (const float*)d_in[19], bfp(oE), bfp(oQ), (float*)(ws + oAcc));

    // decoder
    conv_tmpl<64,32,32,256,256,32,32,3,3,1,1,1,1,true,false,false>
        <<<dim3(128, 16), blk, 576 * 32, stream>>>(bfp(oQ), bfp(wDec0), (const float*)d_in[21], nullptr, bfp(oB1), nullptr);
    P = oB1; T = oB2; O = oB3;
    for (int i = 0; i < 6; i++) {
        conv_tmpl<256,32,32,256,256,32,32,3,3,1,1,1,2,false,false,false>
            <<<dim3(128, 16), blk, 2304 * 32, stream>>>(bfp(P), bfp(wDr1[i]), nullptr, nullptr, bfp(T), nullptr);
        conv_tmpl<256,32,32,256,256,32,32,1,1,1,0,1,0,false,true,false>
            <<<dim3(128, 16), blk, 256 * 32, stream>>>(bfp(T), bfp(wDr2[i]), nullptr, bfp(P), bfp(O), nullptr);
        size_t t2 = P; P = O; O = t2;
    }
    leaky_kernel<<<(8388608 + 255) / 256, 256, 0, stream>>>(bfp(P), 8388608);

    // transpose convs: stride-1 conv over 2x-dilated input, pad 2
    conv_tmpl<256,32,32,128,128,64,64,4,4,1,2,2,1,true,false,false>
        <<<dim3(512, 8), blk, 4096 * 32, stream>>>(bfp(P), bfp(wT0), (const float*)d_in[35], nullptr, bfp(oBig), nullptr);
    conv_tmpl<128,64,64,3,16,128,128,4,4,1,2,2,3,true,false,true>
        <<<dim3(2048, 1), blk, 2048 * 32, stream>>>(bfp(oBig), bfp(wT1), (const float*)d_in[37], nullptr, nullptr, (float*)d_out);

    vq_loss_kernel<<<1, 1, 0, stream>>>((const float*)(ws + oAcc), (float*)d_out + 1572864);
}
